// FastHoloAttention_19301583029009
// MI455X (gfx1250) — compile-verified, matrix operand order checked by reading, not run on hardware
//
#include <hip/hip_runtime.h>
#include <hip/hip_bf16.h>

// ---------------------------------------------------------------------------
// FastHoloAttention for MI455X (gfx1250, wave32, WMMA bf16)
//   B=2, S=2048, D=2048, HARMONICS=64, HEADS=16, Dh=128, N=B*S=4096
// All GEMM work goes through v_wmma_f32_16x16x32_bf16 (f32 accum).
// Attention processes 32 q-rows per wave (2 C tiles) so every K/V fragment
// load feeds 2x the WMMA work -> halves L2 K/V traffic vs 16-row tiles.
// ---------------------------------------------------------------------------

#define NROWS 4096   // B*S
#define DMODEL 2048
#define HARM 64
#define NHEADS 16
#define DHEAD 128
#define SEQ 2048
#define BATCH 2
#define QT 2         // q tiles (16 rows each) per wave in attention

typedef __attribute__((ext_vector_type(16))) __bf16 v16bf;
typedef __attribute__((ext_vector_type(8)))  float  v8f;

union FragU { v16bf v; unsigned u[8]; };

// Load a 16x32 bf16 fragment (A-operand layout per ISA 7.12.2; B-operand uses
// the same loader with the "row" axis = N and contiguous K in memory).
//   lane&15  -> row (M for A / N for B)
//   VGPR i   -> packed K pair k0,k0+1 with
//   k0 = (i/4)*16 + (lane/16)*8 + (i%4)*2
__device__ inline v16bf load_frag16x32(const __hip_bfloat16* base, int ld) {
  const int lane = threadIdx.x & 31;
  const int r    = lane & 15;
  const int khi  = (lane >> 4) << 3;
  FragU f;
#pragma unroll
  for (int i = 0; i < 8; ++i) {
    const int k0 = ((i >> 2) << 4) + khi + ((i & 3) << 1);
    f.u[i] = *reinterpret_cast<const unsigned*>(base + (size_t)r * ld + k0);
  }
  return f.v;
}

__device__ inline v8f wmma_bf16(v16bf a, v16bf b, v8f c) {
  // (neg_a, A, neg_b, B, c_mod, C, reuse_a, reuse_b)
  return __builtin_amdgcn_wmma_f32_16x16x32_bf16(false, a, false, b,
                                                 (short)0, c, false, false);
}

// ---------------------------------------------------------------------------
// Elementwise prep kernels
// ---------------------------------------------------------------------------
__global__ void cvt_f32_bf16(const float* __restrict__ x,
                             __hip_bfloat16* __restrict__ y, int n) {
  int i = blockIdx.x * blockDim.x + threadIdx.x;
  if (i < n) y[i] = __float2bfloat16(x[i]);
}

__global__ void wcos_kernel(const float* __restrict__ phase,
                            const float* __restrict__ amp,
                            __hip_bfloat16* __restrict__ w, int n) {
  int i = blockIdx.x * blockDim.x + threadIdx.x;
  if (i < n) w[i] = __float2bfloat16(amp[i] * __cosf(phase[i]));
}

// ---------------------------------------------------------------------------
// R = X (NROWS x DMODEL) @ basis^T (DMODEL x HARM)  -> R (NROWS x HARM) bf16
// one wave per 16x16 output tile; K loop 2048 in steps of 32
// grid = (NROWS/16, HARM/16), block = 32
// ---------------------------------------------------------------------------
__global__ void __launch_bounds__(32)
resonance_kernel(const __hip_bfloat16* __restrict__ X,
                 const __hip_bfloat16* __restrict__ Bs,  // (HARM x DMODEL)
                 __hip_bfloat16* __restrict__ R) {
  const int n0 = blockIdx.x * 16;
  const int h0 = blockIdx.y * 16;
  v8f c = {};
  for (int k0 = 0; k0 < DMODEL; k0 += 32) {
    v16bf a = load_frag16x32(X  + (size_t)n0 * DMODEL + k0, DMODEL);
    v16bf b = load_frag16x32(Bs + (size_t)h0 * DMODEL + k0, DMODEL);
    c = wmma_bf16(a, b, c);
  }
  const int lane  = threadIdx.x & 31;
  const int col   = lane & 15;
  const int rbase = (lane >> 4) << 3;
#pragma unroll
  for (int r = 0; r < 8; ++r)
    R[(size_t)(n0 + rbase + r) * HARM + h0 + col] = __float2bfloat16(c[r]);
}

// ---------------------------------------------------------------------------
// Out = R (NROWS x HARM) @ w^T (HARM x DMODEL); w stored (DMODEL x HARM)
// mode 0: bf16 store as (bh, s, dh)   [Q, K]
// mode 1: bf16 store as (bh, dh, s)   [V transposed for PV B-fragments]
// mode 2: f32  store row-major (n, o) [final output]
// grid = (NROWS/16, DMODEL/16), block = 32
// ---------------------------------------------------------------------------
__global__ void __launch_bounds__(32)
expand_kernel(const __hip_bfloat16* __restrict__ R,
              const __hip_bfloat16* __restrict__ W,
              void* __restrict__ out, int mode) {
  const int n0 = blockIdx.x * 16;
  const int o0 = blockIdx.y * 16;
  v16bf a0 = load_frag16x32(R + (size_t)n0 * HARM,      HARM);
  v16bf a1 = load_frag16x32(R + (size_t)n0 * HARM + 32, HARM);
  v16bf b0 = load_frag16x32(W + (size_t)o0 * HARM,      HARM);
  v16bf b1 = load_frag16x32(W + (size_t)o0 * HARM + 32, HARM);
  v8f c = {};
  c = wmma_bf16(a0, b0, c);
  c = wmma_bf16(a1, b1, c);
  const int lane  = threadIdx.x & 31;
  const int col   = lane & 15;
  const int rbase = (lane >> 4) << 3;
#pragma unroll
  for (int r = 0; r < 8; ++r) {
    const int n = n0 + rbase + r;
    const int o = o0 + col;
    if (mode == 2) {
      ((float*)out)[(size_t)n * DMODEL + o] = c[r];
    } else {
      const int b = n >> 11, s = n & (SEQ - 1);
      const int h = o >> 7, dh = o & (DHEAD - 1);
      __hip_bfloat16 v = __float2bfloat16(c[r]);
      if (mode == 0)
        ((__hip_bfloat16*)out)[(((size_t)b * NHEADS + h) * SEQ + s) * DHEAD + dh] = v;
      else
        ((__hip_bfloat16*)out)[(((size_t)b * NHEADS + h) * DHEAD + dh) * SEQ + s] = v;
    }
  }
}

// ---------------------------------------------------------------------------
// Flash attention: one wave per (bh, 32-row q block = QT tiles of 16).
// Q,K: (bh, s, dh) bf16 ; V: (bh, dh, s) bf16 ; out: (n, d) bf16 row-major
// Each K/V fragment load is reused by all QT q-tiles.
// grid = (SEQ/(16*QT), BATCH*NHEADS), block = 32
// ---------------------------------------------------------------------------
__global__ void __launch_bounds__(32)
attn_kernel(const __hip_bfloat16* __restrict__ Qb,
            const __hip_bfloat16* __restrict__ Kb,
            const __hip_bfloat16* __restrict__ Vt,
            __hip_bfloat16* __restrict__ Ob) {
  __shared__ __hip_bfloat16 plds[QT][16 * 32];
  const int lane = threadIdx.x & 31;
  const int q0 = blockIdx.x * (16 * QT);
  const int bh = blockIdx.y;               // b*NHEADS + h

  const __hip_bfloat16* Qbase = Qb + ((size_t)bh * SEQ + q0) * DHEAD;
  const __hip_bfloat16* Kbase = Kb + (size_t)bh * SEQ * DHEAD;
  const __hip_bfloat16* Vbase = Vt + (size_t)bh * DHEAD * SEQ;

  v16bf qf[QT][4];
#pragma unroll
  for (int t = 0; t < QT; ++t)
#pragma unroll
    for (int kt = 0; kt < 4; ++kt)
      qf[t][kt] = load_frag16x32(Qbase + (size_t)(t * 16) * DHEAD + kt * 32, DHEAD);

  v8f acc[QT][8];
  float mrow[QT][8], lrow[QT][8];
#pragma unroll
  for (int t = 0; t < QT; ++t) {
#pragma unroll
    for (int j = 0; j < 8; ++j) acc[t][j] = {};
#pragma unroll
    for (int r = 0; r < 8; ++r) { mrow[t][r] = -1e30f; lrow[t][r] = 0.0f; }
  }

  const float scale = 0.08838834764831845f;  // 1/sqrt(128)

  for (int kk0 = 0; kk0 < SEQ; kk0 += 32) {
    // ----- scores: QT*16 q rows x 32 keys; K fragments shared across tiles --
    v8f s0[QT], s1[QT];
#pragma unroll
    for (int t = 0; t < QT; ++t) { s0[t] = {}; s1[t] = {}; }
#pragma unroll
    for (int kt = 0; kt < 4; ++kt) {
      v16bf k0f = load_frag16x32(Kbase + (size_t)kk0 * DHEAD + kt * 32, DHEAD);
      v16bf k1f = load_frag16x32(Kbase + (size_t)(kk0 + 16) * DHEAD + kt * 32, DHEAD);
#pragma unroll
      for (int t = 0; t < QT; ++t) {
        s0[t] = wmma_bf16(qf[t][kt], k0f, s0[t]);
        s1[t] = wmma_bf16(qf[t][kt], k1f, s1[t]);
      }
    }
    // ----- online softmax; row r+8*(lane/16) lives in 16-lane halves -----
    __syncthreads();
#pragma unroll
    for (int t = 0; t < QT; ++t) {
      const int col   = lane & 15;
      const int rbase = (lane >> 4) << 3;
#pragma unroll
      for (int r = 0; r < 8; ++r) {
        float a = s0[t][r] * scale, b = s1[t][r] * scale;
        float x = fmaxf(a, b);
        x = fmaxf(x, __shfl_xor(x, 1, 16));
        x = fmaxf(x, __shfl_xor(x, 2, 16));
        x = fmaxf(x, __shfl_xor(x, 4, 16));
        x = fmaxf(x, __shfl_xor(x, 8, 16));
        float mn    = fmaxf(mrow[t][r], x);
        float alpha = __expf(mrow[t][r] - mn);
        float p0 = __expf(a - mn);
        float p1 = __expf(b - mn);
        float rs = p0 + p1;
        rs += __shfl_xor(rs, 1, 16);
        rs += __shfl_xor(rs, 2, 16);
        rs += __shfl_xor(rs, 4, 16);
        rs += __shfl_xor(rs, 8, 16);
        lrow[t][r] = lrow[t][r] * alpha + rs;
        mrow[t][r] = mn;
#pragma unroll
        for (int j = 0; j < 8; ++j) acc[t][j][r] *= alpha;
        // stage P in LDS (C layout -> A layout re-fragmentation)
        plds[t][(rbase + r) * 32 + col]      = __float2bfloat16(p0);
        plds[t][(rbase + r) * 32 + col + 16] = __float2bfloat16(p1);
      }
    }
    __syncthreads();
    v16bf pf[QT];
#pragma unroll
    for (int t = 0; t < QT; ++t) pf[t] = load_frag16x32(plds[t], 32);
    // ----- O += P @ V : 8 dh tiles of 16; V fragment shared across tiles ---
#pragma unroll
    for (int j = 0; j < 8; ++j) {
      v16bf vf = load_frag16x32(Vbase + (size_t)(j * 16) * SEQ + kk0, SEQ);
#pragma unroll
      for (int t = 0; t < QT; ++t) acc[t][j] = wmma_bf16(pf[t], vf, acc[t][j]);
    }
  }

  // ----- epilogue: normalize, store to (n, d) row-major bf16 -----
  const int b = bh >> 4, h = bh & (NHEADS - 1);
  const int col   = lane & 15;
  const int rbase = (lane >> 4) << 3;
#pragma unroll
  for (int t = 0; t < QT; ++t) {
#pragma unroll
    for (int j = 0; j < 8; ++j) {
#pragma unroll
      for (int r = 0; r < 8; ++r) {
        const float o = acc[t][j][r] / lrow[t][r];
        const size_t n = (size_t)b * SEQ + q0 + t * 16 + rbase + r;
        Ob[n * DMODEL + h * DHEAD + j * 16 + col] = __float2bfloat16(o);
      }
    }
  }
}

// ---------------------------------------------------------------------------
// Host side
// ---------------------------------------------------------------------------
extern "C" void kernel_launch(void* const* d_in, const int* in_sizes, int n_in,
                              void* d_out, int out_size, void* d_ws, size_t ws_size,
                              hipStream_t stream) {
  (void)in_sizes; (void)n_in; (void)out_size; (void)ws_size;
  const float* hs = (const float*)d_in[0];
  const float* basis[4] = { (const float*)d_in[1], (const float*)d_in[4],
                            (const float*)d_in[7], (const float*)d_in[10] };
  const float* phase[4] = { (const float*)d_in[2], (const float*)d_in[5],
                            (const float*)d_in[8], (const float*)d_in[11] };
  const float* amp[4]   = { (const float*)d_in[3], (const float*)d_in[6],
                            (const float*)d_in[9], (const float*)d_in[12] };

  // ---- carve workspace (~85 MB total; L2-resident on MI455X) ----
  char* ws = (char*)d_ws;
  auto carve = [&](size_t bytes) {
    char* p = ws;
    ws += (bytes + 255) & ~(size_t)255;
    return p;
  };
  __hip_bfloat16* Xb = (__hip_bfloat16*)carve((size_t)NROWS * DMODEL * 2);
  __hip_bfloat16* basisb[4];
  __hip_bfloat16* wb[4];
  for (int i = 0; i < 4; ++i) basisb[i] = (__hip_bfloat16*)carve((size_t)HARM * DMODEL * 2);
  for (int i = 0; i < 4; ++i) wb[i]     = (__hip_bfloat16*)carve((size_t)DMODEL * HARM * 2);
  __hip_bfloat16* Rq = (__hip_bfloat16*)carve((size_t)NROWS * HARM * 2);
  __hip_bfloat16* Rk = (__hip_bfloat16*)carve((size_t)NROWS * HARM * 2);
  __hip_bfloat16* Rv = (__hip_bfloat16*)carve((size_t)NROWS * HARM * 2);
  __hip_bfloat16* Ro = (__hip_bfloat16*)carve((size_t)NROWS * HARM * 2);
  __hip_bfloat16* Qb = (__hip_bfloat16*)carve((size_t)NROWS * DMODEL * 2);
  __hip_bfloat16* Kb = (__hip_bfloat16*)carve((size_t)NROWS * DMODEL * 2);
  __hip_bfloat16* Vt = (__hip_bfloat16*)carve((size_t)NROWS * DMODEL * 2);
  __hip_bfloat16* AO = (__hip_bfloat16*)carve((size_t)NROWS * DMODEL * 2);

  const int nX = NROWS * DMODEL;       // 8388608
  const int nP = DMODEL * HARM;        // 131072

  // 1) precision prep
  cvt_f32_bf16<<<(nX + 255) / 256, 256, 0, stream>>>(hs, Xb, nX);
  for (int i = 0; i < 4; ++i) {
    cvt_f32_bf16<<<(nP + 255) / 256, 256, 0, stream>>>(basis[i], basisb[i], nP);
    wcos_kernel<<<(nP + 255) / 256, 256, 0, stream>>>(phase[i], amp[i], wb[i], nP);
  }

  // 2) resonances for q,k,v
  dim3 gR(NROWS / 16, HARM / 16);
  resonance_kernel<<<gR, 32, 0, stream>>>(Xb, basisb[0], Rq);
  resonance_kernel<<<gR, 32, 0, stream>>>(Xb, basisb[1], Rk);
  resonance_kernel<<<gR, 32, 0, stream>>>(Xb, basisb[2], Rv);

  // 3) expand to Q, K (row-major per head) and V (transposed per head)
  dim3 gE(NROWS / 16, DMODEL / 16);
  expand_kernel<<<gE, 32, 0, stream>>>(Rq, wb[0], Qb, 0);
  expand_kernel<<<gE, 32, 0, stream>>>(Rk, wb[1], Kb, 0);
  expand_kernel<<<gE, 32, 0, stream>>>(Rv, wb[2], Vt, 1);

  // 4) flash attention (32 q-rows per wave)
  dim3 gA(SEQ / (16 * QT), BATCH * NHEADS);
  attn_kernel<<<gA, 32, 0, stream>>>(Qb, Kb, Vt, AO);

  // 5) output holo-linear -> f32 d_out
  resonance_kernel<<<gR, 32, 0, stream>>>(AO, basisb[3], Ro);
  expand_kernel<<<gE, 32, 0, stream>>>(Ro, wb[3], d_out, 2);
}
